// Memory_50113678410499
// MI455X (gfx1250) — compile-verified, matrix operand order checked by reading, not run on hardware
//
#include <hip/hip_runtime.h>
#include <hip/hip_bf16.h>

// ---------------- CDNA5 WMMA types ----------------
typedef __attribute__((ext_vector_type(16))) __bf16 v16bf;
typedef __attribute__((ext_vector_type(8)))  __bf16 v8bf;
typedef __attribute__((ext_vector_type(4)))  __bf16 v4bf;
typedef __attribute__((ext_vector_type(8)))  float  v8f;
typedef __attribute__((ext_vector_type(4)))  float  v4f;

#define NQ   524288
#define CDIM 128
#define MDIM 256

// ---------------- LDS layout (ushort units) ----------------
#define W_STRIDE    136   // 128 + 8 pad -> 272B row stride (16B aligned)
#define KEY_STRIDE  136
#define MEMT_STRIDE 264   // 256 + 8 pad -> 528B row stride (16B aligned)
#define SCR_STRIDE  264

#define W_OFF    0
#define KEY_OFF  (W_OFF + 128 * W_STRIDE)        // 17408
#define MEMT_OFF (KEY_OFF + 256 * KEY_STRIDE)    // 52224
#define SCR_OFF  (MEMT_OFF + 128 * MEMT_STRIDE)  // 86016
#define LDS_USHORTS (SCR_OFF + 8 * 16 * SCR_STRIDE) // 119808 ushorts = 239616 B

union Frag {
    v16bf v;
    v8bf  h[2];
    uint4 q[2];
};

// branch-free tanh: 1 - 2*rcp(exp(2x)+1); saturates to +/-1 at both extremes
static __device__ __forceinline__ float fast_tanh(float v) {
    float e = __expf(2.0f * v);
    float r = __builtin_amdgcn_rcpf(e + 1.0f);
    return 1.0f - 2.0f * r;
}

// Gather one 16-bit A/B WMMA fragment from an LDS-resident, K-contiguous array.
// Lane l needs K in [kb+klo, kb+klo+8) and [kb+klo+16, kb+klo+24)  (klo = l<16?0:8),
// at row `row` (M for A-fragments, N-column for B-fragments).
static __device__ __forceinline__ v16bf lds_frag(const unsigned short* smem,
                                                 int baseOff, int row, int stride,
                                                 int kb, int klo) {
    const unsigned short* p = smem + baseOff + row * stride + kb + klo;
    Frag f;
    f.q[0] = *(const uint4*)(p);        // 8 bf16, 16B aligned
    f.q[1] = *(const uint4*)(p + 16);   // next K-chunk (+32B)
    return f.v;
}

// A-fragment straight from global fp32 x (row-major [N][128]):
// four NT b128 loads -> packed bf16 converts.
static __device__ __forceinline__ v16bf afrag_from_x(const float* __restrict__ x,
                                                     int rowBase, int lane, int kb) {
    const int klo = (lane < 16) ? 0 : 8;
    const float* p = x + (size_t)(rowBase + (lane & 15)) * CDIM + kb + klo;
    const v4f a0 = __builtin_nontemporal_load((const v4f*)(p));
    const v4f a1 = __builtin_nontemporal_load((const v4f*)(p + 4));
    const v4f a2 = __builtin_nontemporal_load((const v4f*)(p + 16));
    const v4f a3 = __builtin_nontemporal_load((const v4f*)(p + 20));
    const v8f f0 = __builtin_shufflevector(a0, a1, 0, 1, 2, 3, 4, 5, 6, 7);
    const v8f f1 = __builtin_shufflevector(a2, a3, 0, 1, 2, 3, 4, 5, 6, 7);
    Frag f;
    f.h[0] = __builtin_convertvector(f0, v8bf);
    f.h[1] = __builtin_convertvector(f1, v8bf);
    return f.v;
}

#define WMMA_BF16(A, B, C) \
    __builtin_amdgcn_wmma_f32_16x16x32_bf16(false, (A), false, (B), (short)0, (C), false, false)

extern "C" __global__ void __launch_bounds__(256)
memnet_fused_wmma(const float* __restrict__ x,    // [N,128]
                  const float* __restrict__ W,    // [128,128]
                  const float* __restrict__ bia,  // [128]
                  const float* __restrict__ key,  // [256,128]
                  const float* __restrict__ mem,  // [256,128]
                  float* __restrict__ out,        // [N,128]
                  float* __restrict__ att_out)    // [N,256]
{
    __shared__ __align__(16) unsigned short smem[LDS_USHORTS];

    const int t    = threadIdx.x;
    const int lane = t & 31;
    const int wave = t >> 5;
    const int R    = blockIdx.x * 128 + wave * 16;   // this wave's 16 query rows

    // ---------- stage weights into LDS as bf16 (cooperative, once per block) ----
    // W row-major: B-source for GEMM1 (x @ W^T  => B[k][j] = W[j][k], gather row j, K-contig)
    for (int i = t; i < 128 * 32; i += 256) {
        int r = i >> 5, c = (i & 31) << 2;
        v4bf cv = __builtin_convertvector(*(const v4f*)(W + r * CDIM + c), v4bf);
        *(v4bf*)(&smem[W_OFF + r * W_STRIDE + c]) = cv;   // 8B packed store
    }
    // key row-major: B-source for GEMM2 (xq @ key^T => B[k][m] = key[m][k])
    for (int i = t; i < 256 * 32; i += 256) {
        int r = i >> 5, c = (i & 31) << 2;
        v4bf cv = __builtin_convertvector(*(const v4f*)(key + r * CDIM + c), v4bf);
        *(v4bf*)(&smem[KEY_OFF + r * KEY_STRIDE + c]) = cv;
    }
    // mem transposed: B-source for GEMM3 (att @ mem => B[m][c] = mem[m][c]; need K=m contig)
    for (int i = t; i < 256 * 32; i += 256) {
        int m = i >> 5, c = (i & 31) << 2;
        v4bf cv = __builtin_convertvector(*(const v4f*)(mem + m * CDIM + c), v4bf);
        *(__bf16*)(&smem[MEMT_OFF + (c + 0) * MEMT_STRIDE + m]) = cv[0];
        *(__bf16*)(&smem[MEMT_OFF + (c + 1) * MEMT_STRIDE + m]) = cv[1];
        *(__bf16*)(&smem[MEMT_OFF + (c + 2) * MEMT_STRIDE + m]) = cv[2];
        *(__bf16*)(&smem[MEMT_OFF + (c + 3) * MEMT_STRIDE + m]) = cv[3];
    }
    __syncthreads();

    const int lrow = lane & 15;          // N-column within a 16x16 tile (B/C/D layout)
    const int hi   = lane >> 4;          // half-wave select
    const int klo  = hi ? 8 : 0;         // K sub-offset for A/B fragments
    const int scrBase = SCR_OFF + wave * 16 * SCR_STRIDE;   // per-wave re-layout scratch
    unsigned short* scr = &smem[scrBase];

    // ---------- load A fragments of x (16 rows x 128 K) ----------
    v16bf ax[4];
#pragma unroll
    for (int ks = 0; ks < 4; ++ks) ax[ks] = afrag_from_x(x, R, lane, 32 * ks);

    // ---------- GEMM1: xq = tanh(x @ W^T + b) -> bf16 scratch ----------
#pragma unroll
    for (int jt = 0; jt < 8; ++jt) {
        v8f acc = {};
#pragma unroll
        for (int ks = 0; ks < 4; ++ks) {
            v16bf bW = lds_frag(smem, W_OFF, jt * 16 + lrow, W_STRIDE, 32 * ks, klo);
            acc = WMMA_BF16(ax[ks], bW, acc);
        }
        const float bv = bia[jt * 16 + lrow];
#pragma unroll
        for (int r = 0; r < 8; ++r) {
            // D layout: lanes 0-15 hold M=r, lanes 16-31 hold M=r+8; N = lane%16
            float v = fast_tanh(acc[r] + bv);
            *(__bf16*)(&scr[(r + hi * 8) * SCR_STRIDE + jt * 16 + lrow]) = (__bf16)v;
        }
    }
    // wave-private scratch: per-wave LDS ops are in order; compiler inserts s_wait_dscnt

    // ---------- re-gather xq as A fragments ----------
    v16bf aq[4];
#pragma unroll
    for (int ks = 0; ks < 4; ++ks)
        aq[ks] = lds_frag(smem, scrBase, lrow, SCR_STRIDE, 32 * ks, klo);

    // ---------- GEMM2: logits = xq @ key^T   (16 x 256) ----------
    v8f acc2[16];
#pragma unroll
    for (int mt = 0; mt < 16; ++mt) {
        v8f a = {};
#pragma unroll
        for (int ks = 0; ks < 4; ++ks) {
            v16bf bK = lds_frag(smem, KEY_OFF, mt * 16 + lrow, KEY_STRIDE, 32 * ks, klo);
            a = WMMA_BF16(aq[ks], bK, a);
        }
        acc2[mt] = a;
    }

    // ---------- softmax over M=256 (fp32) ----------
    float inv[8];
#pragma unroll
    for (int r = 0; r < 8; ++r) {
        float mx = acc2[0][r];
#pragma unroll
        for (int mt = 1; mt < 16; ++mt) mx = fmaxf(mx, acc2[mt][r]);
        // reduce across the 16 lanes of this half-wave (same query row set)
#pragma unroll
        for (int s = 8; s >= 1; s >>= 1) mx = fmaxf(mx, __shfl_xor(mx, s, 32));
        float sum = 0.f;
#pragma unroll
        for (int mt = 0; mt < 16; ++mt) {
            float e = __expf(acc2[mt][r] - mx);
            acc2[mt][r] = e;
            sum += e;
        }
#pragma unroll
        for (int s = 8; s >= 1; s >>= 1) sum += __shfl_xor(sum, s, 32);
        inv[r] = __builtin_amdgcn_rcpf(sum);
    }

    // ---------- write att (fp32 output 2, streaming/non-temporal) + stage bf16 ----------
    float* attp = att_out + (size_t)R * MDIM;
#pragma unroll
    for (int mt = 0; mt < 16; ++mt) {
#pragma unroll
        for (int r = 0; r < 8; ++r) {
            const int qr = r + hi * 8;
            const float aw = acc2[mt][r] * inv[r];
            __builtin_nontemporal_store(aw, &attp[(size_t)qr * MDIM + mt * 16 + lrow]);
            *(__bf16*)(&scr[qr * SCR_STRIDE + mt * 16 + lrow]) = (__bf16)aw;
        }
    }

    // ---------- re-gather att as A fragments (K = 256) ----------
    v16bf aa[8];
#pragma unroll
    for (int ks = 0; ks < 8; ++ks)
        aa[ks] = lds_frag(smem, scrBase, lrow, SCR_STRIDE, 32 * ks, klo);

    // ---------- GEMM3: out = att @ mem  (16 x 128) ----------
    float* outp = out + (size_t)R * CDIM;
#pragma unroll
    for (int ct = 0; ct < 8; ++ct) {
        v8f acc3 = {};
#pragma unroll
        for (int ks = 0; ks < 8; ++ks) {
            v16bf bM = lds_frag(smem, MEMT_OFF, ct * 16 + lrow, MEMT_STRIDE, 32 * ks, klo);
            acc3 = WMMA_BF16(aa[ks], bM, acc3);
        }
#pragma unroll
        for (int r = 0; r < 8; ++r)
            __builtin_nontemporal_store(acc3[r],
                &outp[(size_t)(r + hi * 8) * CDIM + ct * 16 + lrow]);
    }
}

extern "C" void kernel_launch(void* const* d_in, const int* in_sizes, int n_in,
                              void* d_out, int out_size, void* d_ws, size_t ws_size,
                              hipStream_t stream) {
    const float* x   = (const float*)d_in[0];
    const float* W   = (const float*)d_in[1];
    const float* b   = (const float*)d_in[2];
    const float* key = (const float*)d_in[3];
    const float* mem = (const float*)d_in[4];
    float* out = (float*)d_out;                         // [N,128] first
    float* att = out + (size_t)NQ * CDIM;               // then [N,256]
    dim3 grid(NQ / 128);                                // 4096 blocks, 128 rows each
    memnet_fused_wmma<<<grid, dim3(256), 0, stream>>>(x, W, b, key, mem, out, att);
    (void)in_sizes; (void)n_in; (void)out_size; (void)d_ws; (void)ws_size;
}
